// GAT_88665304858770
// MI455X (gfx1250) — matrix operand order repaired and checked by reading, not compiled, not run on hardware
//
#include <hip/hip_runtime.h>
#include <hip/hip_bf16.h>

// ---------------------------------------------------------------------------
// 2-layer GAT for gfx1250: bf16 WMMA projections + L2-atomic edge softmax.
// Dims (fixed by reference): IN=256, H=4, F1=32 (H*F1=128), C=40.
// GEMM2 stages its (already-bf16) A operand with global_load_async_to_lds_b128.
// ---------------------------------------------------------------------------

typedef __attribute__((ext_vector_type(16))) __bf16 v16bf;
typedef __attribute__((ext_vector_type(8)))  __bf16 v8bf;
typedef __attribute__((ext_vector_type(8)))  float  v8f;

union Frag16 { v16bf v; v8bf h[2]; };

__device__ __forceinline__ unsigned f2o(float f) {
  unsigned u = __float_as_uint(f);
  return (u & 0x80000000u) ? ~u : (u | 0x80000000u);
}
__device__ __forceinline__ float o2f(unsigned u) {
  return __uint_as_float((u & 0x80000000u) ? (u & 0x7fffffffu) : ~u);
}

// ------------------- GEMM1: (N,256)f32 x (256,128)f32 -> (N,128)f32 --------
// 128-row x 128-col tile per block, 8 waves, each wave: 16 rows x 128 cols.
// f32 -> bf16 conversion fused into LDS staging (features read exactly once).
__global__ __launch_bounds__(256) void gat_gemm1(
    const float* __restrict__ X, const float* __restrict__ W,
    float* __restrict__ Ho, int N)
{
  __shared__ __bf16 sA[128][32];   // [row][k]   8 KB
  __shared__ __bf16 sB[128][32];   // [col][k]   8 KB (transposed W tile)
  const int tid  = threadIdx.x;
  const int wave = tid >> 5, lane = tid & 31;
  const int m = lane & 15, half = lane >> 4;
  const int r0 = blockIdx.x * 128;

  v8f acc[8];
  #pragma unroll
  for (int i = 0; i < 8; ++i)
    #pragma unroll
    for (int j = 0; j < 8; ++j) acc[i][j] = 0.0f;

  const int arow = tid >> 1;
  const int akc  = (tid & 1) << 4;
  const int grow = r0 + arow;

  for (int k0 = 0; k0 < 256; k0 += 32) {
    // Stage A tile (convert f32 -> bf16): each thread 16 contiguous floats.
    if (grow < N) {
      const float* p = X + (size_t)grow * 256 + k0 + akc;
      #pragma unroll
      for (int i = 0; i < 16; ++i) sA[arow][akc + i] = (__bf16)p[i];
    } else {
      #pragma unroll
      for (int i = 0; i < 16; ++i) sA[arow][akc + i] = (__bf16)0.0f;
    }
    // Stage B tile transposed: coalesced global read, [col][k] LDS layout.
    #pragma unroll
    for (int i = 0; i < 16; ++i) {
      int e = tid + (i << 8);            // 0..4095
      int col = e & 127, k = e >> 7;     // col fast-varying -> coalesced
      sB[col][k] = (__bf16)W[(size_t)(k0 + k) * 128 + col];
    }
    __syncthreads();

    Frag16 a;   // ISA 16-bit A 16x32 layout
    a.h[0] = *(const v8bf*)&sA[wave * 16 + m][half * 8];
    a.h[1] = *(const v8bf*)&sA[wave * 16 + m][16 + half * 8];
    #pragma unroll
    for (int ct = 0; ct < 8; ++ct) {
      Frag16 b; // B 32x16 layout matching A: K 8*half..+7, 16+8*half..+7
      b.h[0] = *(const v8bf*)&sB[ct * 16 + m][half * 8];
      b.h[1] = *(const v8bf*)&sB[ct * 16 + m][16 + half * 8];
      acc[ct] = __builtin_amdgcn_wmma_f32_16x16x32_bf16(
          false, a.v, false, b.v, (short)0, acc[ct], false, false);
    }
    __syncthreads();
  }
  // C/D layout: VGPR r, lane l -> row r + 8*(l/16), col l%16
  #pragma unroll
  for (int ct = 0; ct < 8; ++ct)
    #pragma unroll
    for (int r = 0; r < 8; ++r) {
      int row = r0 + wave * 16 + r + 8 * half;
      if (row < N) Ho[(size_t)row * 128 + ct * 16 + m] = acc[ct][r];
    }
}

// ------------------- GEMM2: (N,128)bf16 x (128,40)f32 -> (N,40)f32 ---------
// W2 staged fully into LDS once; A tiles staged with async LDS DMA (bf16 in
// global already, no conversion needed). Rows >= N read in-ws poison and are
// discarded by the store guard.
__global__ __launch_bounds__(256) void gat_gemm2(
    const __bf16* __restrict__ X, const float* __restrict__ W,
    float* __restrict__ Ho, int N)
{
  __shared__ __bf16 sA[128][32];   // [row][k]      8 KB
  __shared__ __bf16 sB[48][128];   // [col][k-full] 12 KB, cols padded 40->48
  const int tid  = threadIdx.x;
  const int wave = tid >> 5, lane = tid & 31;
  const int m = lane & 15, half = lane >> 4;
  const int r0 = blockIdx.x * 128;

  v8f acc[3];
  #pragma unroll
  for (int i = 0; i < 3; ++i)
    #pragma unroll
    for (int j = 0; j < 8; ++j) acc[i][j] = 0.0f;

  // Stage entire W2 (128x40 f32 -> bf16, padded to 48 cols) once.
  #pragma unroll
  for (int i = 0; i < 24; ++i) {       // 48*128 = 6144 = 24*256
    int e = tid + (i << 8);
    int col = e >> 7, k = e & 127;
    float v = (col < 40) ? W[(size_t)k * 40 + col] : 0.0f;
    sB[col][k] = (__bf16)v;
  }

  for (int k0 = 0; k0 < 128; k0 += 32) {
    // Async-DMA the 128x32 bf16 A tile: 512 chunks of 16B, 2 per thread.
    #pragma unroll
    for (int i = 0; i < 2; ++i) {
      int c = tid + (i << 8);          // 0..511
      int row = c >> 2, part = c & 3;
      unsigned laddr = (unsigned)(size_t)&sA[row][part * 8];
      const __bf16* gp = X + (size_t)(r0 + row) * 128 + k0 + part * 8;
      asm volatile("global_load_async_to_lds_b128 %0, %1, off"
                   :: "v"(laddr), "v"(gp) : "memory");
    }
    asm volatile("s_wait_asynccnt 0x0" ::: "memory");
    __syncthreads();

    Frag16 a;
    a.h[0] = *(const v8bf*)&sA[wave * 16 + m][half * 8];
    a.h[1] = *(const v8bf*)&sA[wave * 16 + m][16 + half * 8];
    #pragma unroll
    for (int ct = 0; ct < 3; ++ct) {
      Frag16 b;
      b.h[0] = *(const v8bf*)&sB[ct * 16 + m][k0 + half * 8];
      b.h[1] = *(const v8bf*)&sB[ct * 16 + m][k0 + 16 + half * 8];
      acc[ct] = __builtin_amdgcn_wmma_f32_16x16x32_bf16(
          false, a.v, false, b.v, (short)0, acc[ct], false, false);
    }
    __syncthreads();
  }
  #pragma unroll
  for (int ct = 0; ct < 3; ++ct)
    #pragma unroll
    for (int r = 0; r < 8; ++r) {
      int row = r0 + wave * 16 + r + 8 * half;
      int col = ct * 16 + m;
      if (row < N && col < 40) Ho[(size_t)row * 40 + col] = acc[ct][r];
    }
}

// ------------------- per-node attention halves ------------------------------
__global__ void gat_nodeatt1(const float* __restrict__ Hf,
                             const float* __restrict__ al, const float* __restrict__ ar,
                             float* __restrict__ el, float* __restrict__ er, int N)
{
  int t = blockIdx.x * blockDim.x + threadIdx.x;
  if (t >= N * 4) return;
  int n = t >> 2, hd = t & 3;
  const float* hr  = Hf + (size_t)n * 128 + hd * 32;
  const float* alp = al + hd * 32;
  const float* arp = ar + hd * 32;
  float sl = 0.f, sr = 0.f;
  #pragma unroll
  for (int f = 0; f < 32; ++f) { float v = hr[f]; sl += v * alp[f]; sr += v * arp[f]; }
  el[t] = sl; er[t] = sr;
}

__global__ void gat_nodeatt2(const float* __restrict__ Hf,
                             const float* __restrict__ al, const float* __restrict__ ar,
                             float* __restrict__ el, float* __restrict__ er, int N)
{
  int n = blockIdx.x * blockDim.x + threadIdx.x;
  if (n >= N) return;
  const float* hr = Hf + (size_t)n * 40;
  float sl = 0.f, sr = 0.f;
  #pragma unroll
  for (int c = 0; c < 40; ++c) { float v = hr[c]; sl += v * al[c]; sr += v * ar[c]; }
  el[n] = sl; er[n] = sr;
}

// ------------------- init kernels ------------------------------------------
__global__ void gat_init_att(unsigned* __restrict__ mm, float* __restrict__ ss, int n)
{
  int t = blockIdx.x * blockDim.x + threadIdx.x;
  if (t >= n) return;
  mm[t] = f2o(-__builtin_inff());
  ss[t] = 0.f;
}

__global__ void gat_init_bias(float* __restrict__ out, const float* __restrict__ b,
                              int total, int Fdim)
{
  int t = blockIdx.x * blockDim.x + threadIdx.x;
  if (t >= total) return;
  out[t] = b[t % Fdim];
}

// ------------------- edge softmax passes -----------------------------------
template<int HH>
__global__ void gat_edge_max(const int* __restrict__ src, const int* __restrict__ dst,
                             const float* __restrict__ el, const float* __restrict__ er,
                             float* __restrict__ esc, unsigned* __restrict__ mmax, int E)
{
  int t = blockIdx.x * blockDim.x + threadIdx.x;
  if (t >= E * HH) return;
  int e = t / HH, hd = t - e * HH;
  int s = src[e], d = dst[e];
  float x = el[s * HH + hd] + er[d * HH + hd];
  x = x > 0.f ? x : 0.2f * x;                 // LeakyReLU(0.2)
  esc[t] = x;
  atomicMax(&mmax[d * HH + hd], f2o(x));
}

template<int HH>
__global__ void gat_edge_exp(const int* __restrict__ dst,
                             float* __restrict__ esc, const unsigned* __restrict__ mmax,
                             float* __restrict__ ssum, int E)
{
  int t = blockIdx.x * blockDim.x + threadIdx.x;
  if (t >= E * HH) return;
  int e = t / HH, hd = t - e * HH;
  int d = dst[e];
  float mv = o2f(mmax[d * HH + hd]);
  float ee = __expf(esc[t] - mv);
  esc[t] = ee;
  atomicAdd(&ssum[d * HH + hd], ee);
}

// ------------------- weighted scatter-add aggregation ----------------------
// 32 threads per edge, one float4 each: h[src] row read as one contiguous
// 512B burst per edge; 4 global_atomic_add_f32 per thread into out[dst].
__global__ void gat_edge_aggr1(const int* __restrict__ src, const int* __restrict__ dst,
                               const float* __restrict__ esc, const float* __restrict__ ssum,
                               const float* __restrict__ Hf, float* __restrict__ out, int E)
{
  long long t = (long long)blockIdx.x * blockDim.x + threadIdx.x;
  if (t >= (long long)E * 32) return;
  int e  = (int)(t >> 5);
  int j4 = ((int)t & 31) << 2;        // feature offset 0..124
  int hd = j4 >> 5;
  int s = src[e], d = dst[e];
  float alpha = esc[(size_t)e * 4 + hd] / ssum[(size_t)d * 4 + hd];
  const float4 hv = *(const float4*)(Hf + (size_t)s * 128 + j4);
  float* orow = out + (size_t)d * 128 + j4;
  atomicAdd(&orow[0], alpha * hv.x);
  atomicAdd(&orow[1], alpha * hv.y);
  atomicAdd(&orow[2], alpha * hv.z);
  atomicAdd(&orow[3], alpha * hv.w);
}

// 16 threads per edge covering 40 cols (10 active float4 lanes of 16).
__global__ void gat_edge_aggr2(const int* __restrict__ src, const int* __restrict__ dst,
                               const float* __restrict__ esc, const float* __restrict__ ssum,
                               const float* __restrict__ Hf, float* __restrict__ out, int E)
{
  long long t = (long long)blockIdx.x * blockDim.x + threadIdx.x;
  if (t >= (long long)E * 16) return;
  int e  = (int)(t >> 4);
  int j4 = ((int)t & 15) << 2;        // 0..60
  if (j4 >= 40) return;
  int s = src[e], d = dst[e];
  float alpha = esc[e] / ssum[d];
  const float4 hv = *(const float4*)(Hf + (size_t)s * 40 + j4);   // stride 160B, 16B aligned
  float* orow = out + (size_t)d * 40 + j4;
  atomicAdd(&orow[0], alpha * hv.x);
  atomicAdd(&orow[1], alpha * hv.y);
  atomicAdd(&orow[2], alpha * hv.z);
  atomicAdd(&orow[3], alpha * hv.w);
}

// ------------------- ELU + f32->bf16 for layer-2 input ---------------------
__global__ void gat_elu_bf16(const float* __restrict__ in, __bf16* __restrict__ out, int n)
{
  int t = blockIdx.x * blockDim.x + threadIdx.x;
  if (t >= n) return;
  float v = in[t];
  out[t] = (__bf16)(v > 0.f ? v : (__expf(v) - 1.f));
}

// ---------------------------------------------------------------------------
extern "C" void kernel_launch(void* const* d_in, const int* in_sizes, int n_in,
                              void* d_out, int out_size, void* d_ws, size_t ws_size,
                              hipStream_t stream)
{
  const float* X   = (const float*)d_in[0];
  const int*   src = (const int*)d_in[1];
  const int*   dst = (const int*)d_in[2];
  const float* W1  = (const float*)d_in[3];
  const float* al1 = (const float*)d_in[4];
  const float* ar1 = (const float*)d_in[5];
  const float* b1  = (const float*)d_in[6];
  const float* W2  = (const float*)d_in[7];
  const float* al2 = (const float*)d_in[8];
  const float* ar2 = (const float*)d_in[9];
  const float* b2  = (const float*)d_in[10];
  (void)n_in; (void)out_size; (void)ws_size;

  const int N = in_sizes[0] / 256;
  const int E = in_sizes[1];

  char* ws = (char*)d_ws;
  size_t off = 0;
  auto alloc = [&](size_t bytes) -> void* {
    void* p = ws + off;
    off = (off + bytes + 255) & ~(size_t)255;
    return p;
  };
  float*    h1  = (float*)   alloc((size_t)N * 128 * 4);
  float*    el1 = (float*)   alloc((size_t)N * 4 * 4);
  float*    er1 = (float*)   alloc((size_t)N * 4 * 4);
  unsigned* m1  = (unsigned*)alloc((size_t)N * 4 * 4);
  float*    s1  = (float*)   alloc((size_t)N * 4 * 4);
  float*    e1  = (float*)   alloc((size_t)E * 4 * 4);
  float*    o1  = (float*)   alloc((size_t)N * 128 * 4);
  __bf16*   x2  = (__bf16*)  alloc((size_t)N * 128 * 2);
  float*    h2  = (float*)   alloc((size_t)N * 40 * 4);
  float*    el2 = (float*)   alloc((size_t)N * 4);
  float*    er2 = (float*)   alloc((size_t)N * 4);
  unsigned* m2  = (unsigned*)alloc((size_t)N * 4);
  float*    s2  = (float*)   alloc((size_t)N * 4);
  float*    e2  = (float*)   alloc((size_t)E * 4);

  const int B = 256;
  auto cd = [](long long a, long long b) { return (unsigned)((a + b - 1) / b); };

  // ---- layer 1 ----
  gat_gemm1<<<cd(N, 128), B, 0, stream>>>(X, W1, h1, N);
  gat_nodeatt1<<<cd((long long)N * 4, B), B, 0, stream>>>(h1, al1, ar1, el1, er1, N);
  gat_init_att<<<cd((long long)N * 4, B), B, 0, stream>>>(m1, s1, N * 4);
  gat_init_bias<<<cd((long long)N * 128, B), B, 0, stream>>>(o1, b1, N * 128, 128);
  gat_edge_max<4><<<cd((long long)E * 4, B), B, 0, stream>>>(src, dst, el1, er1, e1, m1, E);
  gat_edge_exp<4><<<cd((long long)E * 4, B), B, 0, stream>>>(dst, e1, m1, s1, E);
  gat_edge_aggr1<<<cd((long long)E * 32, B), B, 0, stream>>>(src, dst, e1, s1, h1, o1, E);
  gat_elu_bf16<<<cd((long long)N * 128, B), B, 0, stream>>>(o1, x2, N * 128);

  // ---- layer 2 ----
  gat_gemm2<<<cd(N, 128), B, 0, stream>>>(x2, W2, h2, N);
  gat_nodeatt2<<<cd(N, B), B, 0, stream>>>(h2, al2, ar2, el2, er2, N);
  gat_init_att<<<cd(N, B), B, 0, stream>>>(m2, s2, N);
  gat_init_bias<<<cd((long long)N * 40, B), B, 0, stream>>>((float*)d_out, b2, N * 40, 40);
  gat_edge_max<1><<<cd(E, B), B, 0, stream>>>(src, dst, el2, er2, e2, m2, E);
  gat_edge_exp<1><<<cd(E, B), B, 0, stream>>>(dst, e2, m2, s2, E);
  gat_edge_aggr2<<<cd((long long)E * 16, B), B, 0, stream>>>(src, dst, e2, s2, h2, (float*)d_out, E);
}
